// RSSM_7808250544890
// MI455X (gfx1250) — matrix operand order repaired and checked by reading, not compiled, not run on hardware
//
#include <hip/hip_runtime.h>
#include <hip/hip_bf16.h>

// ---------------- problem constants ----------------
#define BB   128
#define SS   256
#define ENC  512
#define HD   512
#define SD   64
#define HD3  1536   // 3*HD
#define GINP 96     // GRU input 66 padded to 96 (mult of 32 for WMMA K)
#define NWG  16
#define EPSV 1e-5f

typedef __attribute__((ext_vector_type(16))) __bf16 v16bf;
typedef __attribute__((ext_vector_type(8)))  __bf16 v8bf;
typedef __attribute__((ext_vector_type(8)))  float  v8f;

// ---------------- workspace layout (bytes, all 256B aligned) ----------------
constexpr size_t OFF_BAR  = 0;                                      // 256 B barrier
constexpr size_t OFF_WIHT = 256;                                    // 96x1536 bf16
constexpr size_t OFF_WHHT = OFF_WIHT + (size_t)96*1536*2;           // 512x1536
constexpr size_t OFF_PW1T = OFF_WHHT + (size_t)512*1536*2;          // 512x256
constexpr size_t OFF_PW2T = OFF_PW1T + (size_t)512*256*2;           // 256x128
constexpr size_t OFF_PW3T = OFF_PW2T + (size_t)256*128*2;           // 128x128
constexpr size_t OFF_PW4T = OFF_PW3T + (size_t)128*128*2;
constexpr size_t OFF_PWFT = OFF_PW4T + (size_t)128*128*2;
constexpr size_t OFF_SW1T = OFF_PWFT + (size_t)128*128*2;           // 1024x256
constexpr size_t OFF_SW2T = OFF_SW1T + (size_t)1024*256*2;          // 256x128
constexpr size_t OFF_SW3T = OFF_SW2T + (size_t)256*128*2;
constexpr size_t OFF_SW4T = OFF_SW3T + (size_t)128*128*2;
constexpr size_t OFF_SWFT = OFF_SW4T + (size_t)128*128*2;
constexpr size_t OFF_HF   = OFF_SWFT + (size_t)128*128*2;           // h f32 128x512
constexpr size_t OFF_HB   = OFF_HF   + (size_t)128*512*4;           // h bf16
constexpr size_t OFF_XG   = OFF_HB   + (size_t)128*512*2;           // gru in bf16 128x96
constexpr size_t OFF_XB   = OFF_XG   + (size_t)128*96*2;            // post in bf16 128x1024
constexpr size_t OFF_PA   = OFF_XB   + (size_t)128*1024*2;          // prior act A 128x256 bf16
constexpr size_t OFF_PB   = OFF_PA   + (size_t)128*256*2;           // prior act B
constexpr size_t OFF_QA   = OFF_PB   + (size_t)128*256*2;           // post act A
constexpr size_t OFF_QB   = OFF_QA   + (size_t)128*256*2;           // post act B
constexpr size_t OFF_Y1   = OFF_QB   + (size_t)128*256*2;           // gi f32 128x1536
constexpr size_t OFF_Y2   = OFF_Y1   + (size_t)128*1536*4;          // gh f32 128x1536
constexpr size_t OFF_YP   = OFF_Y2   + (size_t)128*1536*4;          // prior pre-BN 128x256 f32
constexpr size_t OFF_YQ   = OFF_YP   + (size_t)128*256*4;           // post pre-BN
constexpr size_t OFF_ZH   = OFF_YQ   + (size_t)128*256*4;           // z_hat f32 128x128
constexpr size_t OFF_Z    = OFF_ZH   + (size_t)128*128*4;           // z f32 128x128
constexpr size_t WS_NEED  = OFF_Z    + (size_t)128*128*4;

// output layout (floats)
constexpr size_t OUT_H    = 0;
constexpr size_t OUT_SMP  = OUT_H    + (size_t)BB*SS*HD;
constexpr size_t OUT_ZMU  = OUT_SMP  + (size_t)BB*SS*SD;
constexpr size_t OUT_ZLS  = OUT_ZMU  + (size_t)BB*SS*SD;
constexpr size_t OUT_ZHMU = OUT_ZLS  + (size_t)BB*SS*SD;
constexpr size_t OUT_ZHLS = OUT_ZHMU + (size_t)BB*SS*SD;

// ---------------- helpers ----------------
__device__ __forceinline__ unsigned short f2bf(float f) {
  unsigned u = __float_as_uint(f);
  unsigned r = (u + 0x7FFFu + ((u >> 16) & 1u)) >> 16;   // round-to-nearest-even
  return (unsigned short)r;
}

__device__ __forceinline__ void grid_sync(unsigned int* bar) {
  __syncthreads();
  if (threadIdx.x == 0) {
    volatile unsigned int* vbar = (volatile unsigned int*)bar;
    __threadfence();
    unsigned int gen = vbar[1];
    unsigned int old = atomicAdd(bar, 1u);
    if (old == (unsigned)(NWG - 1)) {
      vbar[0] = 0u;
      __threadfence();
      atomicAdd(bar + 1, 1u);
    } else {
      while (vbar[1] == gen) { __builtin_amdgcn_s_sleep(1); }
    }
    __threadfence();
  }
  __syncthreads();
}

// GEMM: Y[128xN] = A[128xK](bf16,row-major,lda=K) * Bt[KxN](bf16,row-major) + bias
// Each wave computes 16x16 tiles with v_wmma_f32_16x16x32_bf16.
__device__ __forceinline__ void gemm_bf16(
    const unsigned short* __restrict__ A, int lda,
    const unsigned short* __restrict__ Bt,
    const float* __restrict__ bias,
    float* __restrict__ Y, int K, int N,
    int wave, int nwaves, int lane)
{
  const int numTiles = 8 * (N >> 4);        // M=128 -> 8 row blocks
  const int r  = lane & 15;
  const int hi = lane >> 4;
  for (int t = wave; t < numTiles; t += nwaves) {
    const int m0 = (t & 7) << 4;
    const int n0 = (t >> 3) << 4;
    v8f acc = {};
    const unsigned short* arow = A + (size_t)(m0 + r) * lda;
    for (int k0 = 0; k0 < K; k0 += 32) {
      // A 16x32 layout: lanes 0-15 rows m, K=[k0+hi*8..+7] then [k0+16+hi*8..+7]
      v8bf alo = *(const v8bf*)(const void*)(arow + k0 + hi * 8);
      v8bf ahi = *(const v8bf*)(const void*)(arow + k0 + 16 + hi * 8);
      v16bf a = __builtin_shufflevector(alo, ahi, 0,1,2,3,4,5,6,7,8,9,10,11,12,13,14,15);
      // B 32x16 layout: lane (hi*16+r) holds column n0+r, K=[k0+hi*8..+7] then [k0+16+hi*8..+7]
      v16bf b;
      for (int s = 0; s < 16; ++s) {
        const int kk = (s < 8) ? (hi * 8 + s) : (16 + hi * 8 + (s - 8));
        b[s] = __builtin_bit_cast(__bf16, Bt[(size_t)(k0 + kk) * N + n0 + r]);
      }
      acc = __builtin_amdgcn_wmma_f32_16x16x32_bf16(false, a, false, b, (short)0, acc,
                                                    false, false);
    }
    const float bc = bias ? bias[n0 + r] : 0.0f;
#pragma unroll
    for (int v = 0; v < 8; ++v) {
      // C/D layout: VGPR v -> row m0+v (lanes 0-15) / m0+v+8 (lanes 16-31), col n0+r
      Y[(size_t)(m0 + v + hi * 8) * N + n0 + r] = acc[v] + bc;
    }
  }
}

// BatchNorm(training stats over B=128) + ReLU + cvt to bf16, one thread per column
__device__ __forceinline__ void bn_relu(
    const float* __restrict__ Y, const float* __restrict__ g,
    const float* __restrict__ be, unsigned short* __restrict__ Out,
    int N, int gtid, int nth)
{
  for (int c = gtid; c < N; c += nth) {
    float s = 0.f, ss = 0.f;
    for (int m = 0; m < BB; ++m) { float v = Y[(size_t)m * N + c]; s += v; ss += v * v; }
    float mu  = s * (1.0f / BB);
    float var = ss * (1.0f / BB) - mu * mu;
    float a   = rsqrtf(var + EPSV) * g[c];
    float b0  = be[c] - mu * a;
    for (int m = 0; m < BB; ++m) {
      float v = Y[(size_t)m * N + c] * a + b0;
      Out[(size_t)m * N + c] = f2bf(v > 0.f ? v : 0.f);
    }
  }
}

// ---------------- init kernel: bar reset + weight transpose/convert to bf16 ----------------
struct InitParams {
  const float* W[12];
  unsigned short* Wt[12];
  int N[12], K[12], Kp[12];
  unsigned int* bar;
};

__global__ void rssm_init_k(InitParams p) {
  const int gtid = blockIdx.x * blockDim.x + threadIdx.x;
  const int nth  = gridDim.x * blockDim.x;
  if (gtid < 64) p.bar[gtid] = 0u;
  for (int j = 0; j < 12; ++j) {
    const float* W = p.W[j];
    unsigned short* Wt = p.Wt[j];
    const int N = p.N[j], K = p.K[j], Kp = p.Kp[j];
    const int tot = Kp * N;
    for (int i = gtid; i < tot; i += nth) {
      int k = i / N, n = i - k * N;                 // Wt[k*N+n] = W[n*K+k]
      Wt[i] = (k < K) ? f2bf(W[(size_t)n * K + k]) : (unsigned short)0;
    }
  }
}

// ---------------- main persistent kernel ----------------
struct MainParams {
  const float *emb, *act, *noise, *bih, *bhh;
  const float *pb1,*pg1,*pbe1,*pb2,*pg2,*pbe2,*pb3,*pg3,*pbe3,*pb4,*pg4,*pbe4,*pbf;
  const float *sb1,*sg1,*sbe1,*sb2,*sg2,*sbe2,*sb3,*sg3,*sbe3,*sb4,*sg4,*sbe4,*sbf;
  float* out;
  char* ws;
};

__global__ __launch_bounds__(256, 1) void rssm_main_k(MainParams p) {
  char* ws = p.ws;
  unsigned int*   bar    = (unsigned int*)(ws + OFF_BAR);
  const unsigned short* WIHT = (const unsigned short*)(ws + OFF_WIHT);
  const unsigned short* WHHT = (const unsigned short*)(ws + OFF_WHHT);
  const unsigned short* PW1T = (const unsigned short*)(ws + OFF_PW1T);
  const unsigned short* PW2T = (const unsigned short*)(ws + OFF_PW2T);
  const unsigned short* PW3T = (const unsigned short*)(ws + OFF_PW3T);
  const unsigned short* PW4T = (const unsigned short*)(ws + OFF_PW4T);
  const unsigned short* PWFT = (const unsigned short*)(ws + OFF_PWFT);
  const unsigned short* SW1T = (const unsigned short*)(ws + OFF_SW1T);
  const unsigned short* SW2T = (const unsigned short*)(ws + OFF_SW2T);
  const unsigned short* SW3T = (const unsigned short*)(ws + OFF_SW3T);
  const unsigned short* SW4T = (const unsigned short*)(ws + OFF_SW4T);
  const unsigned short* SWFT = (const unsigned short*)(ws + OFF_SWFT);
  float*          hF   = (float*)(ws + OFF_HF);
  unsigned short* hB   = (unsigned short*)(ws + OFF_HB);
  unsigned short* xG   = (unsigned short*)(ws + OFF_XG);
  unsigned short* xB   = (unsigned short*)(ws + OFF_XB);
  unsigned short* pA   = (unsigned short*)(ws + OFF_PA);
  unsigned short* pBuf = (unsigned short*)(ws + OFF_PB);
  unsigned short* qA   = (unsigned short*)(ws + OFF_QA);
  unsigned short* qB   = (unsigned short*)(ws + OFF_QB);
  float* y1 = (float*)(ws + OFF_Y1);
  float* y2 = (float*)(ws + OFF_Y2);
  float* yp = (float*)(ws + OFF_YP);
  float* yq = (float*)(ws + OFF_YQ);
  float* zh = (float*)(ws + OFF_ZH);
  float* zz = (float*)(ws + OFF_Z);

  const int tid    = threadIdx.x;
  const int gtid   = blockIdx.x * 256 + tid;
  const int nth    = NWG * 256;
  const int wave   = gtid >> 5;
  const int nwaves = NWG * 8;
  const int lane   = tid & 31;

  for (int t = 0; t < SS; ++t) {
    if (t > 0) {
      // ---- GRU GEMMs: gi = x_gru @ WihT + bih ; gh = h @ WhhT + bhh ----
      gemm_bf16(xG, GINP, WIHT, p.bih, y1, GINP, HD3, wave, nwaves, lane);
      gemm_bf16(hB, HD,   WHHT, p.bhh, y2, HD,   HD3, wave, nwaves, lane);
      grid_sync(bar);
      // ---- GRU gates + h update + build post input [h | emb_t] ----
      for (int i = gtid; i < BB * HD; i += nth) {
        int m = i >> 9, j = i & 511;
        float ir = y1[(size_t)m*HD3 + j],        hr = y2[(size_t)m*HD3 + j];
        float iz = y1[(size_t)m*HD3 + HD + j],   hz = y2[(size_t)m*HD3 + HD + j];
        float in = y1[(size_t)m*HD3 + 2*HD + j], hn = y2[(size_t)m*HD3 + 2*HD + j];
        float rg = 1.f / (1.f + __expf(-(ir + hr)));
        float zg = 1.f / (1.f + __expf(-(iz + hz)));
        float ng = tanhf(in + rg * hn);
        float hv = (1.f - zg) * ng + zg * hF[i];
        hF[i] = hv;
        unsigned short hb = f2bf(hv);
        hB[i] = hb;
        p.out[OUT_H + ((size_t)m * SS + t) * HD + j] = hv;
        xB[(size_t)m * 1024 + j] = hb;
        xB[(size_t)m * 1024 + ENC + j] = f2bf(p.emb[((size_t)m * SS + t) * ENC + j]);
      }
    } else {
      // ---- t==0: h = 0, build post input ----
      for (int i = gtid; i < BB * HD; i += nth) {
        int m = i >> 9, j = i & 511;
        hF[i] = 0.f; hB[i] = 0;
        p.out[OUT_H + ((size_t)m * SS) * HD + j] = 0.f;
        xB[(size_t)m * 1024 + j] = 0;
        xB[(size_t)m * 1024 + ENC + j] = f2bf(p.emb[((size_t)m * SS) * ENC + j]);
      }
    }
    grid_sync(bar);

    // ---- prior L1 (K=512,N=256) + post L1 (K=1024,N=256) ----
    gemm_bf16(hB, HD,   PW1T, p.pb1, yp, HD,   256, wave, nwaves, lane);
    gemm_bf16(xB, 1024, SW1T, p.sb1, yq, 1024, 256, wave, nwaves, lane);
    grid_sync(bar);
    bn_relu(yp, p.pg1, p.pbe1, pA, 256, gtid, nth);
    bn_relu(yq, p.sg1, p.sbe1, qA, 256, gtid, nth);
    grid_sync(bar);
    // ---- L2 (K=256,N=128) ----
    gemm_bf16(pA, 256, PW2T, p.pb2, yp, 256, 128, wave, nwaves, lane);
    gemm_bf16(qA, 256, SW2T, p.sb2, yq, 256, 128, wave, nwaves, lane);
    grid_sync(bar);
    bn_relu(yp, p.pg2, p.pbe2, pBuf, 128, gtid, nth);
    bn_relu(yq, p.sg2, p.sbe2, qB,   128, gtid, nth);
    grid_sync(bar);
    // ---- L3 (128x128) ----
    gemm_bf16(pBuf, 128, PW3T, p.pb3, yp, 128, 128, wave, nwaves, lane);
    gemm_bf16(qB,   128, SW3T, p.sb3, yq, 128, 128, wave, nwaves, lane);
    grid_sync(bar);
    bn_relu(yp, p.pg3, p.pbe3, pA, 128, gtid, nth);
    bn_relu(yq, p.sg3, p.sbe3, qA, 128, gtid, nth);
    grid_sync(bar);
    // ---- L4 (128x128) ----
    gemm_bf16(pA, 128, PW4T, p.pb4, yp, 128, 128, wave, nwaves, lane);
    gemm_bf16(qA, 128, SW4T, p.sb4, yq, 128, 128, wave, nwaves, lane);
    grid_sync(bar);
    bn_relu(yp, p.pg4, p.pbe4, pBuf, 128, gtid, nth);
    bn_relu(yq, p.sg4, p.sbe4, qB,   128, gtid, nth);
    grid_sync(bar);
    // ---- final Linear (128x128), no BN ----
    gemm_bf16(pBuf, 128, PWFT, p.pbf, zh, 128, 128, wave, nwaves, lane);
    gemm_bf16(qB,   128, SWFT, p.sbf, zz, 128, 128, wave, nwaves, lane);
    grid_sync(bar);

    // ---- outputs + reparameterized sample + GRU input for t+1 ----
    for (int i = gtid; i < BB * SD; i += nth) {
      int m = i >> 6, j = i & 63;
      float zmu = zz[(size_t)m * 128 + j];
      float zls = zz[(size_t)m * 128 + SD + j];
      float nz  = p.noise[((size_t)m * SS + t) * SD + j];
      float smp = zmu + __expf(zls) * nz;
      size_t o = ((size_t)m * SS + t) * SD + j;
      p.out[OUT_SMP  + o] = smp;
      p.out[OUT_ZMU  + o] = zmu;
      p.out[OUT_ZLS  + o] = zls;
      p.out[OUT_ZHMU + o] = zh[(size_t)m * 128 + j];
      p.out[OUT_ZHLS + o] = zh[(size_t)m * 128 + SD + j];
      xG[(size_t)m * GINP + j] = f2bf(smp);          // sample part of next GRU input
    }
    for (int i = gtid; i < BB * 32; i += nth) {      // action part + zero pad (cols 64..95)
      int m = i >> 5, j = i & 31;
      unsigned short v = 0;
      if (j < 2) v = f2bf(p.act[((size_t)m * SS + t) * 2 + j]);
      xG[(size_t)m * GINP + SD + j] = v;
    }
    grid_sync(bar);
  }
}

// ---------------- host launch ----------------
extern "C" void kernel_launch(void* const* d_in, const int* in_sizes, int n_in,
                              void* d_out, int out_size, void* d_ws, size_t ws_size,
                              hipStream_t stream) {
  if (ws_size < WS_NEED) return;
  auto F = [&](int i) { return (const float*)d_in[i]; };
  char* ws = (char*)d_ws;

  InitParams ip;
  const int    widx[12] = {3, 4, 7, 11, 15, 19, 23, 25, 29, 33, 37, 41};
  const size_t woff[12] = {OFF_WIHT, OFF_WHHT, OFF_PW1T, OFF_PW2T, OFF_PW3T, OFF_PW4T,
                           OFF_PWFT, OFF_SW1T, OFF_SW2T, OFF_SW3T, OFF_SW4T, OFF_SWFT};
  const int wN[12]  = {1536, 1536, 256, 128, 128, 128, 128, 256, 128, 128, 128, 128};
  const int wK[12]  = {66,   512,  512, 256, 128, 128, 128, 1024, 256, 128, 128, 128};
  const int wKp[12] = {96,   512,  512, 256, 128, 128, 128, 1024, 256, 128, 128, 128};
  for (int j = 0; j < 12; ++j) {
    ip.W[j]  = F(widx[j]);
    ip.Wt[j] = (unsigned short*)(ws + woff[j]);
    ip.N[j] = wN[j]; ip.K[j] = wK[j]; ip.Kp[j] = wKp[j];
  }
  ip.bar = (unsigned int*)(ws + OFF_BAR);
  rssm_init_k<<<128, 256, 0, stream>>>(ip);

  MainParams mp;
  mp.emb = F(0); mp.act = F(1); mp.noise = F(2);
  mp.bih = F(5); mp.bhh = F(6);
  mp.pb1 = F(8);  mp.pg1 = F(9);  mp.pbe1 = F(10);
  mp.pb2 = F(12); mp.pg2 = F(13); mp.pbe2 = F(14);
  mp.pb3 = F(16); mp.pg3 = F(17); mp.pbe3 = F(18);
  mp.pb4 = F(20); mp.pg4 = F(21); mp.pbe4 = F(22);
  mp.pbf = F(24);
  mp.sb1 = F(26); mp.sg1 = F(27); mp.sbe1 = F(28);
  mp.sb2 = F(30); mp.sg2 = F(31); mp.sbe2 = F(32);
  mp.sb3 = F(34); mp.sg3 = F(35); mp.sbe3 = F(36);
  mp.sb4 = F(38); mp.sg4 = F(39); mp.sbe4 = F(40);
  mp.sbf = F(42);
  mp.out = (float*)d_out;
  mp.ws  = ws;
  rssm_main_k<<<NWG, 256, 0, stream>>>(mp);
}
